// MolecularGNN_23115513987316
// MI455X (gfx1250) — compile-verified, matrix operand order checked by reading, not run on hardware
//
#include <hip/hip_runtime.h>

#define N_NODES   100000
#define N_EDGES   1600000
#define NODE_DIM  6
#define HID       256
#define OUTD      512
#define N_GRAPHS  512
#define N_LAYERS  4
#define BN_EPS    1e-5f

typedef __attribute__((ext_vector_type(16))) __bf16 v16bf;
typedef __attribute__((ext_vector_type(8)))  float  v8f;

// ---------------------------------------------------------------- helpers ---
__device__ __forceinline__ float4 bnrelu4(float4 a, float4 s, float4 t) {
    float4 r;
    r.x = fmaxf(fmaf(a.x, s.x, t.x), 0.f);
    r.y = fmaxf(fmaf(a.y, s.y, t.y), 0.f);
    r.z = fmaxf(fmaf(a.z, s.z, t.z), 0.f);
    r.w = fmaxf(fmaf(a.w, s.w, t.w), 0.f);
    return r;
}

// ------------------------------------------------------------ degree/norm ---
__global__ void k_deg_init(float* deg, int n) {
    int i = blockIdx.x * blockDim.x + threadIdx.x;
    if (i < n) deg[i] = 1.0f;                    // self-loop contribution
}
__global__ void k_deg_edges(const int* __restrict__ ei, float* deg, int e) {
    int i = blockIdx.x * blockDim.x + threadIdx.x;
    if (i < e) atomicAdd(&deg[ei[e + i]], 1.0f); // dst = ei[1][i]
}
__global__ void k_dinv(const float* __restrict__ deg, float* dinv, int n) {
    int i = blockIdx.x * blockDim.x + threadIdx.x;
    if (i < n) dinv[i] = rsqrtf(deg[i]);
}

// ------------------------------------------- pack fp32 W[K,N] -> WMMA bf16 ---
// Per 32x16 (KxN) tile: lane l = (n&15) + 16*hi holds 16 contiguous bf16.
__global__ void k_pack_w(const float* __restrict__ W, __bf16* __restrict__ P,
                         int K, int N) {
    int idx = blockIdx.x * blockDim.x + threadIdx.x;
    if (idx >= K * N) return;
    int k = idx / N, n = idx % N;
    int tk = k >> 5, tn = n >> 4;
    int kin = k & 31, n16 = n & 15;
    int hi = (kin >> 3) & 1;
    int kk = kin - (hi << 3);                    // {0..7} U {16..23}
    int j  = (kk < 8) ? kk : (kk - 8);
    int lane = n16 + (hi << 4);
    int Ntiles = N >> 4;
    size_t dst = (((size_t)tk * Ntiles + tn) << 9) + ((size_t)lane << 4) + j;
    P[dst] = (__bf16)W[idx];
}

// -------------------------------------------------------------- node proj ---
__global__ __launch_bounds__(256) void k_node_proj(
        const float* __restrict__ x, const float* __restrict__ W,
        const float* __restrict__ b, float* __restrict__ h) {
    __shared__ float xs[8];
    int n = blockIdx.x, t = threadIdx.x;
    if (t < NODE_DIM) xs[t] = x[n * NODE_DIM + t];
    __syncthreads();
    float s = b[t];
#pragma unroll
    for (int k = 0; k < NODE_DIM; ++k) s = fmaf(xs[k], W[k * HID + t], s);
    h[(size_t)n * HID + t] = s;
}

// -------------------------------------------------- WMMA bf16 GEMM (f32 C) ---
// C[M,N] = op(A)[M,K] @ Bpacked[K,N] (+bias)(+relu).  op(A)=relu(A*scale+shift)
// when BN (fused BN of previous layer). One wave -> 16x64 strip.
template <bool BN>
__global__ __launch_bounds__(256) void k_gemm_bf16(
        const float* __restrict__ A, const __bf16* __restrict__ Bp,
        float* __restrict__ C, const float* __restrict__ scale,
        const float* __restrict__ shift, const float* __restrict__ bias,
        int M, int N, int K, int relu_out) {
    const int lane    = threadIdx.x & 31;
    const int nstrips = N >> 6;
    const int total   = (M >> 4) * nstrips;
    const int wid     = blockIdx.x * 8 + (threadIdx.x >> 5);
    if (wid >= total) return;                    // whole-wave exit: EXEC stays full
    const int mtile = wid / nstrips;
    const int strip = wid % nstrips;
    const int nBase = strip << 6;
    const int hi    = lane >> 4;
    const int n16   = lane & 15;
    const int mrow  = (mtile << 4) + n16;
    const int Ntiles = N >> 4;
    const int ksteps = K >> 5;

    v8f c0 = {}, c1 = {}, c2 = {}, c3 = {};
    const float* arow = A + (size_t)mrow * K;

    for (int kt = 0; kt < ksteps; ++kt) {
        const int kb = (kt << 5) + (hi << 3);
        float4 a0 = *(const float4*)(arow + kb);
        float4 a1 = *(const float4*)(arow + kb + 4);
        float4 a2 = *(const float4*)(arow + kb + 16);
        float4 a3 = *(const float4*)(arow + kb + 20);
        __builtin_prefetch(arow + kb + 32, 0, 1);          // global_prefetch_b8
        if (BN) {                                          // fused BN + ReLU
            a0 = bnrelu4(a0, *(const float4*)(scale + kb),      *(const float4*)(shift + kb));
            a1 = bnrelu4(a1, *(const float4*)(scale + kb + 4),  *(const float4*)(shift + kb + 4));
            a2 = bnrelu4(a2, *(const float4*)(scale + kb + 16), *(const float4*)(shift + kb + 16));
            a3 = bnrelu4(a3, *(const float4*)(scale + kb + 20), *(const float4*)(shift + kb + 20));
        }
        v16bf av;                                          // native f32->bf16 casts
        av[0]  = (__bf16)a0.x; av[1]  = (__bf16)a0.y; av[2]  = (__bf16)a0.z; av[3]  = (__bf16)a0.w;
        av[4]  = (__bf16)a1.x; av[5]  = (__bf16)a1.y; av[6]  = (__bf16)a1.z; av[7]  = (__bf16)a1.w;
        av[8]  = (__bf16)a2.x; av[9]  = (__bf16)a2.y; av[10] = (__bf16)a2.z; av[11] = (__bf16)a2.w;
        av[12] = (__bf16)a3.x; av[13] = (__bf16)a3.y; av[14] = (__bf16)a3.z; av[15] = (__bf16)a3.w;

        const __bf16* bt =
            Bp + (((size_t)kt * Ntiles + (nBase >> 4)) << 9) + ((size_t)lane << 4);
        v16bf b0 = *(const v16bf*)(bt);
        v16bf b1 = *(const v16bf*)(bt + 512);
        v16bf b2 = *(const v16bf*)(bt + 1024);
        v16bf b3 = *(const v16bf*)(bt + 1536);

        c0 = __builtin_amdgcn_wmma_f32_16x16x32_bf16(false, av, false, b0, (short)0, c0, false, false);
        c1 = __builtin_amdgcn_wmma_f32_16x16x32_bf16(false, av, false, b1, (short)0, c1, false, false);
        c2 = __builtin_amdgcn_wmma_f32_16x16x32_bf16(false, av, false, b2, (short)0, c2, false, false);
        c3 = __builtin_amdgcn_wmma_f32_16x16x32_bf16(false, av, false, b3, (short)0, c3, false, false);
    }

    const int col0 = nBase + n16;
    float bb0 = 0.f, bb1 = 0.f, bb2 = 0.f, bb3 = 0.f;
    if (bias) { bb0 = bias[col0]; bb1 = bias[col0 + 16]; bb2 = bias[col0 + 32]; bb3 = bias[col0 + 48]; }
#pragma unroll
    for (int v = 0; v < 8; ++v) {
        int row = (mtile << 4) + (hi << 3) + v;
        float* crow = C + (size_t)row * N;
        float x0 = c0[v] + bb0, x1 = c1[v] + bb1, x2 = c2[v] + bb2, x3 = c3[v] + bb3;
        if (relu_out) { x0 = fmaxf(x0, 0.f); x1 = fmaxf(x1, 0.f); x2 = fmaxf(x2, 0.f); x3 = fmaxf(x3, 0.f); }
        crow[col0]      = x0;
        crow[col0 + 16] = x1;
        crow[col0 + 32] = x2;
        crow[col0 + 48] = x3;
    }
}

// ------------------------------------------- scatter: bias + self-loop init ---
__global__ __launch_bounds__(256) void k_acc_init(
        const float* __restrict__ hw, const float* __restrict__ dinv,
        const float* __restrict__ bc, float* __restrict__ acc) {
    int n = blockIdx.x, t = threadIdx.x;
    float d = dinv[n];
    size_t i = (size_t)n * HID + t;
    acc[i] = fmaf(hw[i], d * d, bc[t]);
}

// one edge per wave; 8 channels per lane; f32 atomic scatter-add
__global__ __launch_bounds__(256) void k_edge_scatter(
        const int* __restrict__ ei, const float* __restrict__ hw,
        const float* __restrict__ dinv, float* __restrict__ acc, int E) {
    int lane = threadIdx.x & 31;
    int e = blockIdx.x * 8 + (threadIdx.x >> 5);
    if (e >= E) return;
    int s = ei[e], d = ei[E + e];
    float w = dinv[s] * dinv[d];
    const float* hs = hw + (size_t)s * HID + lane * 8;
    float* ad = acc + (size_t)d * HID + lane * 8;
    float4 v0 = *(const float4*)(hs);
    float4 v1 = *(const float4*)(hs + 4);
    atomicAdd(ad + 0, v0.x * w); atomicAdd(ad + 1, v0.y * w);
    atomicAdd(ad + 2, v0.z * w); atomicAdd(ad + 3, v0.w * w);
    atomicAdd(ad + 4, v1.x * w); atomicAdd(ad + 5, v1.y * w);
    atomicAdd(ad + 6, v1.z * w); atomicAdd(ad + 7, v1.w * w);
}

// ------------------------------------------------------------- batch norm ---
__global__ void k_bn_zero(float* s, float* s2) { s[threadIdx.x] = 0.f; s2[threadIdx.x] = 0.f; }

__global__ __launch_bounds__(256) void k_bn_stats(
        const float* __restrict__ acc, float* __restrict__ sum,
        float* __restrict__ sumsq, int rows) {
    int t = threadIdx.x;
    size_t r0 = (size_t)blockIdx.x * rows;
    float s = 0.f, s2 = 0.f;
    for (int r = 0; r < rows; ++r) {
        float v = acc[(r0 + r) * HID + t];
        s += v; s2 = fmaf(v, v, s2);
    }
    atomicAdd(&sum[t], s); atomicAdd(&sumsq[t], s2);
}

__global__ void k_bn_final(const float* __restrict__ sum, const float* __restrict__ sumsq,
                           const float* __restrict__ gamma, const float* __restrict__ beta,
                           float* __restrict__ scale, float* __restrict__ shift, float invN) {
    int t = threadIdx.x;
    float mu  = sum[t] * invN;
    float var = fmaxf(sumsq[t] * invN - mu * mu, 0.f);
    float sc  = rsqrtf(var + BN_EPS) * gamma[t];
    scale[t] = sc;
    shift[t] = beta[t] - mu * sc;
}

// ---------------------------------------------------------------- pooling ---
__global__ void k_pool_init(float* gsum, unsigned int* gmax, float* cnt) {
    int i = blockIdx.x * blockDim.x + threadIdx.x;
    gsum[i] = 0.f; gmax[i] = 0u;                 // ReLU => features >= 0
    if (i < N_GRAPHS) cnt[i] = 0.f;
}

__global__ __launch_bounds__(256) void k_nodefeat_pool(
        const float* __restrict__ acc, const float* __restrict__ scale,
        const float* __restrict__ shift, const int* __restrict__ batch,
        float* __restrict__ nf, float* __restrict__ gsum,
        unsigned int* __restrict__ gmax, float* __restrict__ cnt) {
    int n = blockIdx.x, t = threadIdx.x;
    int g = batch[n];
    size_t i = (size_t)n * HID + t;
    float v = fmaxf(fmaf(acc[i], scale[t], shift[t]), 0.f);
    nf[i] = v;
    atomicAdd(&gsum[g * HID + t], v);
    atomicMax(&gmax[g * HID + t], __float_as_uint(v));  // monotone for v >= 0
    if (t == 0) atomicAdd(&cnt[g], 1.0f);
}

__global__ void k_pool_final(const float* __restrict__ gsum,
                             const unsigned int* __restrict__ gmax,
                             const float* __restrict__ cnt, float* __restrict__ pooled) {
    int g = blockIdx.x, t = threadIdx.x;
    float c = cnt[g];
    pooled[(size_t)g * (2 * HID) + t]       = gsum[g * HID + t] / c;
    pooled[(size_t)g * (2 * HID) + HID + t] = __uint_as_float(gmax[g * HID + t]);
}

// ------------------------------------------------------------------- host ---
static inline char* bump(char*& p, size_t bytes) {
    char* r = p;
    p += (bytes + 255) & ~(size_t)255;
    return r;
}

extern "C" void kernel_launch(void* const* d_in, const int* in_sizes, int n_in,
                              void* d_out, int out_size, void* d_ws, size_t ws_size,
                              hipStream_t stream) {
    (void)in_sizes; (void)n_in; (void)out_size; (void)ws_size;
    const float* x     = (const float*)d_in[0];
    const int*   ei    = (const int*)d_in[1];
    const int*   batch = (const int*)d_in[2];
    const float* Wp    = (const float*)d_in[3];
    const float* bp    = (const float*)d_in[4];
    const float* Wc[4] = {(const float*)d_in[5], (const float*)d_in[7],
                          (const float*)d_in[9], (const float*)d_in[11]};
    const float* bc[4] = {(const float*)d_in[6], (const float*)d_in[8],
                          (const float*)d_in[10], (const float*)d_in[12]};
    const float* gam[4] = {(const float*)d_in[13], (const float*)d_in[15],
                           (const float*)d_in[17], (const float*)d_in[19]};
    const float* bet[4] = {(const float*)d_in[14], (const float*)d_in[16],
                           (const float*)d_in[18], (const float*)d_in[20]};
    // d_in[21..24] = attn params (unused by reference computation)
    const float* W1 = (const float*)d_in[25];
    const float* b1 = (const float*)d_in[26];
    const float* W2 = (const float*)d_in[27];
    const float* b2 = (const float*)d_in[28];

    float* mol = (float*)d_out;                          // [512, 512]
    float* nf  = (float*)d_out + (size_t)N_GRAPHS * OUTD; // [100000, 256]

    char* p = (char*)d_ws;
    float*        deg    = (float*)bump(p, (size_t)N_NODES * 4);
    float*        dinv   = (float*)bump(p, (size_t)N_NODES * 4);
    float*        bnsum  = (float*)bump(p, HID * 4);
    float*        bnsq   = (float*)bump(p, HID * 4);
    float*        scale  = (float*)bump(p, HID * 4);
    float*        shift  = (float*)bump(p, HID * 4);
    float*        cnt    = (float*)bump(p, N_GRAPHS * 4);
    float*        gsum   = (float*)bump(p, (size_t)N_GRAPHS * HID * 4);
    unsigned int* gmax   = (unsigned int*)bump(p, (size_t)N_GRAPHS * HID * 4);
    float*        pooled = (float*)bump(p, (size_t)N_GRAPHS * 2 * HID * 4);
    float*        t1     = (float*)bump(p, (size_t)N_GRAPHS * HID * 4);
    __bf16*       pWc    = (__bf16*)bump(p, (size_t)4 * HID * HID * 2);
    __bf16*       pW1    = (__bf16*)bump(p, (size_t)2 * HID * HID * 2);
    __bf16*       pW2    = (__bf16*)bump(p, (size_t)HID * OUTD * 2);
    float*        bufH   = (float*)bump(p, (size_t)N_NODES * HID * 4);
    float*        bufT   = (float*)bump(p, (size_t)N_NODES * HID * 4);

    // GCN symmetric norm
    k_deg_init<<<(N_NODES + 255) / 256, 256, 0, stream>>>(deg, N_NODES);
    k_deg_edges<<<(N_EDGES + 255) / 256, 256, 0, stream>>>(ei, deg, N_EDGES);
    k_dinv<<<(N_NODES + 255) / 256, 256, 0, stream>>>(deg, dinv, N_NODES);

    // weight packing (bf16, WMMA-B register layout)
    for (int l = 0; l < N_LAYERS; ++l)
        k_pack_w<<<(HID * HID + 255) / 256, 256, 0, stream>>>(Wc[l], pWc + (size_t)l * HID * HID, HID, HID);
    k_pack_w<<<(2 * HID * HID + 255) / 256, 256, 0, stream>>>(W1, pW1, 2 * HID, HID);
    k_pack_w<<<(HID * OUTD + 255) / 256, 256, 0, stream>>>(W2, pW2, HID, OUTD);

    // h0 = x @ Wp + bp
    k_node_proj<<<N_NODES, 256, 0, stream>>>(x, Wp, bp, bufH);

    // GCN layers: GEMM (fused prev BN+ReLU on A) -> self-loop+bias init -> edge scatter -> BN stats
    for (int l = 0; l < N_LAYERS; ++l) {
        int waves  = (N_NODES / 16) * (HID / 64);
        int blocks = (waves + 7) / 8;
        if (l == 0)
            k_gemm_bf16<false><<<blocks, 256, 0, stream>>>(
                bufH, pWc, bufT, nullptr, nullptr, nullptr, N_NODES, HID, HID, 0);
        else
            k_gemm_bf16<true><<<blocks, 256, 0, stream>>>(
                bufH, pWc + (size_t)l * HID * HID, bufT, scale, shift,
                nullptr, N_NODES, HID, HID, 0);
        k_acc_init<<<N_NODES, 256, 0, stream>>>(bufT, dinv, bc[l], bufH);
        k_edge_scatter<<<N_EDGES / 8, 256, 0, stream>>>(ei, bufT, dinv, bufH, N_EDGES);
        k_bn_zero<<<1, 256, 0, stream>>>(bnsum, bnsq);
        k_bn_stats<<<400, 256, 0, stream>>>(bufH, bnsum, bnsq, N_NODES / 400);
        k_bn_final<<<1, 256, 0, stream>>>(bnsum, bnsq, gam[l], bet[l], scale, shift, 1.0f / N_NODES);
    }

    // node features (final BN+ReLU) + fused mean/max pooling
    k_pool_init<<<(N_GRAPHS * HID) / 256, 256, 0, stream>>>(gsum, gmax, cnt);
    k_nodefeat_pool<<<N_NODES, 256, 0, stream>>>(bufH, scale, shift, batch, nf, gsum, gmax, cnt);
    k_pool_final<<<N_GRAPHS, 256, 0, stream>>>(gsum, gmax, cnt, pooled);

    // head MLP: t1 = relu(pooled @ W1 + b1); mol = t1 @ W2 + b2
    {
        int waves = (N_GRAPHS / 16) * (HID / 64);
        k_gemm_bf16<false><<<(waves + 7) / 8, 256, 0, stream>>>(
            pooled, pW1, t1, nullptr, nullptr, b1, N_GRAPHS, HID, 2 * HID, 1);
    }
    {
        int waves = (N_GRAPHS / 16) * (OUTD / 64);
        k_gemm_bf16<false><<<(waves + 7) / 8, 256, 0, stream>>>(
            t1, pW2, mol, nullptr, nullptr, b2, N_GRAPHS, OUTD, HID, 0);
    }
}
